// GNNLayer_28217935135265
// MI455X (gfx1250) — matrix operand order repaired and checked by reading, not compiled, not run on hardware
//
#include <hip/hip_runtime.h>
#include <cstdint>

typedef __attribute__((ext_vector_type(16))) _Float16 v16h;
typedef __attribute__((ext_vector_type(8)))  _Float16 v8h;
typedef __attribute__((ext_vector_type(8)))  float    v8f;

#define N_NODES 50000
#define N_EDGES 800000
#define NODE_C  128
#define EDGE_C  64

#define EDGE_W_BYTES  (320 * 128 * 2 + 128 * 128 * 2)   // 114688 (W1 | W2 contiguous)
#define NODE_W_BYTES  (256 * 128 * 2 + 128 * 128 * 2)   // 98304  (W1 | W2 contiguous)
#define HTILE_BYTES   (8 * 16 * 128 * 2)                // 32768  (8 waves x 16x128 f16)

__device__ __forceinline__ float silu_f(float x) {
    return x / (1.0f + __expf(-x));
}

__device__ __forceinline__ void atomic_add_f32(float* p, float v) {
    __hip_atomic_fetch_add(p, v, __ATOMIC_RELAXED, __HIP_MEMORY_SCOPE_AGENT);
}

// CDNA5 async global->LDS copy (16B per lane), tracked by ASYNCcnt.
__device__ __forceinline__ void async_copy_b128(void* lptr, const void* gptr) {
    unsigned lds_off =
        (unsigned)(uintptr_t)(__attribute__((address_space(3))) void*)lptr;
    asm volatile("global_load_async_to_lds_b128 %0, %1, off"
                 :: "v"(lds_off), "v"((unsigned long long)(uintptr_t)gptr)
                 : "memory");
}

__device__ __forceinline__ void wait_async0() {
    asm volatile("s_wait_asynccnt 0" ::: "memory");
}

// Stage `bytes` (multiple of 4096) from global into LDS, 256 threads.
__device__ __forceinline__ void stage_weights(char* lds, const char* g,
                                              unsigned bytes) {
    for (unsigned off = threadIdx.x * 16u; off < bytes; off += 256u * 16u)
        async_copy_b128(lds + off, g + off);
    wait_async0();
    __syncthreads();
}

// ---------------------------------------------------------------------------
// Zero the aggregation buffer (scratch is not re-poisoned between replays).
// ---------------------------------------------------------------------------
__global__ void zero_f32_kernel(float* __restrict__ p, long long n) {
    long long i = (long long)blockIdx.x * blockDim.x + threadIdx.x;
    long long stride = (long long)gridDim.x * blockDim.x;
    for (; i < n; i += stride) p[i] = 0.0f;
}

// ---------------------------------------------------------------------------
// Pre-swizzle a row-major f32 weight [K][N] into f16 WMMA B-fragments.
// Fragment layout: frag = kc*(N/16) + nt ; within frag: lane*16 + elem.
// B 32x16 f16 layout: half h = lane/16, elem e -> K = (e/8)*16 + h*8 + e%8;
//                     column n = nt*16 + lane%16.
// ---------------------------------------------------------------------------
__global__ void swizzle_weight_kernel(const float* __restrict__ src,
                                      _Float16* __restrict__ dst,
                                      int K, int N) {
    int total = K * N;
    int nTiles = N >> 4;
    for (int t = blockIdx.x * blockDim.x + threadIdx.x; t < total;
         t += gridDim.x * blockDim.x) {
        int f      = t >> 9;        // fragment id (512 halves each)
        int within = t & 511;
        int lane   = within >> 4;
        int elem   = within & 15;
        int kc = f / nTiles;
        int nt = f % nTiles;
        int n  = nt * 16 + (lane & 15);
        int h  = lane >> 4;
        int v  = elem >> 1;
        int s  = elem & 1;
        int k  = kc * 32 + (elem >> 3) * 16 + h * 8 + (elem & 7); (void)v; (void)s;
        dst[t] = (_Float16)src[k * N + n];
    }
}

// Select segment pointer inside the virtual concatenated row.
__device__ __forceinline__ const float* seg3(const float* s0, const float* s1,
                                             const float* s2, int k) {
    return (k < 128) ? (s0 + k) : (k < 256) ? (s1 + (k - 128)) : (s2 + (k - 256));
}
__device__ __forceinline__ const float* seg2(const float* s0, const float* s1,
                                             int k) {
    return (k < 128) ? (s0 + k) : (s1 + (k - 128));
}

// Build 16-bit A fragment (16x32) from two 8-float runs.
// A layout: lane m = lane%16, half h = lane/16:
//   a[0..7]  <- K = kc*32 + h*8 + {0..7}
//   a[8..15] <- K = kc*32 + 16 + h*8 + {0..7}
__device__ __forceinline__ v16h make_a_frag(const float* p0, const float* p1) {
    v8f r0 = *(const v8f*)p0;
    v8f r1 = *(const v8f*)p1;
    v16h a;
#pragma unroll
    for (int i = 0; i < 8; ++i) {
        a[i]     = (_Float16)r0[i];
        a[i + 8] = (_Float16)r1[i];
    }
    return a;
}

// B fragment from LDS-resident swizzled weights (2 x ds_load_b128).
__device__ __forceinline__ v16h lds_b_frag(const _Float16* p) {
    v8h b0 = *(const v8h*)p;
    v8h b1 = *(const v8h*)(p + 8);
    v16h b;
#pragma unroll
    for (int i = 0; i < 8; ++i) { b[i] = b0[i]; b[i + 8] = b1[i]; }
    return b;
}

// ---------------------------------------------------------------------------
// Edge MLP (persistent blocks, weights LDS-resident):
//   h = silu([x_s|x_r|x_e] @ W1 + b1) ; msg = silu(h @ W2 + b2)
//   atomic scatter-add msg into aggr[receiver].
// ---------------------------------------------------------------------------
__global__ void __launch_bounds__(256)
edge_mlp_kernel(const float* __restrict__ x_nodes,
                const float* __restrict__ x_edges,
                const long long* __restrict__ sender,
                const long long* __restrict__ receiver,
                const _Float16* __restrict__ wAll,   // W1 frags | W2 frags
                const float* __restrict__ b1, const float* __restrict__ b2,
                float* __restrict__ aggr, int numTiles) {
    extern __shared__ char smem[];
    _Float16* ldsW1 = (_Float16*)smem;                         // 40960 halves
    _Float16* ldsW2 = (_Float16*)(smem + 320 * 128 * 2);       // 16384 halves
    _Float16* ldsH  = (_Float16*)(smem + EDGE_W_BYTES);        // 8*2048 halves

    stage_weights(smem, (const char*)wAll, EDGE_W_BYTES);

    const int lane   = threadIdx.x & 31;
    const int waveId = threadIdx.x >> 5;
    const int m = lane & 15;
    const int h = lane >> 4;
    const int tileStride = gridDim.x * 8;
    _Float16* hTile = ldsH + waveId * 2048;
    const v8f vzero = {};

    for (int tile = blockIdx.x * 8 + waveId; tile < numTiles;
         tile += tileStride) {
        const long long e = (long long)tile * 16 + m;
        const float* rowS = x_nodes + (long long)sender[e]   * NODE_C;
        const float* rowR = x_nodes + (long long)receiver[e] * NODE_C;
        const float* rowE = x_edges + e * EDGE_C;

        // prefetch next tile's edge-feature row (the only big HBM stream)
        if (tile + tileStride < numTiles)
            __builtin_prefetch(x_edges +
                               ((long long)(tile + tileStride) * 16 + m) * EDGE_C,
                               0, 3);

        v8f acc[8];
#pragma unroll
        for (int nt = 0; nt < 8; ++nt) acc[nt] = vzero;

        // GEMM1: K = 320 (10 chunks of 32), B frags from LDS
        for (int kc = 0; kc < 10; ++kc) {
            const int kb = kc * 32;
            v16h a = make_a_frag(seg3(rowS, rowR, rowE, kb + h * 8),
                                 seg3(rowS, rowR, rowE, kb + 16 + h * 8));
            const _Float16* wbase = ldsW1 + kc * 8 * 512 + lane * 16;
#pragma unroll
            for (int nt = 0; nt < 8; ++nt) {
                v16h b = lds_b_frag(wbase + nt * 512);
                acc[nt] = __builtin_amdgcn_wmma_f32_16x16x32_f16(
                    false, a, false, b, (short)0, acc[nt], false, false);
            }
        }

        // bias + SiLU, stage 16x128 h-tile in this wave's LDS slab
#pragma unroll
        for (int nt = 0; nt < 8; ++nt) {
            const int n = nt * 16 + m;
            const float bias = b1[n];
#pragma unroll
            for (int r = 0; r < 8; ++r) {
                float x = acc[nt][r] + bias;
                hTile[(r + 8 * h) * 128 + n] = (_Float16)silu_f(x);
            }
        }

        // GEMM2: K = 128 (4 chunks)
        v8f acc2[8];
#pragma unroll
        for (int nt = 0; nt < 8; ++nt) acc2[nt] = vzero;

        for (int kc = 0; kc < 4; ++kc) {
            const _Float16* pa = hTile + m * 128 + kc * 32 + h * 8;
            v8h a0 = *(const v8h*)pa;
            v8h a1 = *(const v8h*)(pa + 16);
            v16h a;
#pragma unroll
            for (int i = 0; i < 8; ++i) { a[i] = a0[i]; a[i + 8] = a1[i]; }
            const _Float16* wbase = ldsW2 + kc * 8 * 512 + lane * 16;
#pragma unroll
            for (int nt = 0; nt < 8; ++nt) {
                v16h b = lds_b_frag(wbase + nt * 512);
                acc2[nt] = __builtin_amdgcn_wmma_f32_16x16x32_f16(
                    false, a, false, b, (short)0, acc2[nt], false, false);
            }
        }

        // bias + SiLU, atomic scatter-add to receivers (L2-resident target)
        long long rcvOff[8];
#pragma unroll
        for (int r = 0; r < 8; ++r)
            rcvOff[r] = receiver[(long long)tile * 16 + r + 8 * h] * NODE_C;

#pragma unroll
        for (int nt = 0; nt < 8; ++nt) {
            const int n = nt * 16 + m;
            const float bias = b2[n];
#pragma unroll
            for (int r = 0; r < 8; ++r) {
                float v = silu_f(acc2[nt][r] + bias);
                atomic_add_f32(&aggr[rcvOff[r] + n], v);
            }
        }
    }
}

// ---------------------------------------------------------------------------
// Node MLP (persistent blocks, weights LDS-resident):
//   u = silu([x | aggr] @ W1 + b1) ; out = u @ W2 + b2
// ---------------------------------------------------------------------------
__global__ void __launch_bounds__(256)
node_mlp_kernel(const float* __restrict__ x_nodes,
                const float* __restrict__ aggr,
                const _Float16* __restrict__ wAll,   // W1 frags | W2 frags
                const float* __restrict__ b1, const float* __restrict__ b2,
                float* __restrict__ out, int numTiles) {
    extern __shared__ char smem[];
    _Float16* ldsW1 = (_Float16*)smem;                         // 32768 halves
    _Float16* ldsW2 = (_Float16*)(smem + 256 * 128 * 2);       // 16384 halves
    _Float16* ldsH  = (_Float16*)(smem + NODE_W_BYTES);

    stage_weights(smem, (const char*)wAll, NODE_W_BYTES);

    const int lane   = threadIdx.x & 31;
    const int waveId = threadIdx.x >> 5;
    const int m = lane & 15;
    const int h = lane >> 4;
    _Float16* hTile = ldsH + waveId * 2048;
    const v8f vzero = {};

    for (int tile = blockIdx.x * 8 + waveId; tile < numTiles;
         tile += gridDim.x * 8) {
        const long long node = (long long)tile * 16 + m;
        const float* rowX = x_nodes + node * NODE_C;
        const float* rowA = aggr    + node * NODE_C;

        v8f acc[8];
#pragma unroll
        for (int nt = 0; nt < 8; ++nt) acc[nt] = vzero;

        // GEMM1: K = 256 (8 chunks)
        for (int kc = 0; kc < 8; ++kc) {
            const int kb = kc * 32;
            v16h a = make_a_frag(seg2(rowX, rowA, kb + h * 8),
                                 seg2(rowX, rowA, kb + 16 + h * 8));
            const _Float16* wbase = ldsW1 + kc * 8 * 512 + lane * 16;
#pragma unroll
            for (int nt = 0; nt < 8; ++nt) {
                v16h b = lds_b_frag(wbase + nt * 512);
                acc[nt] = __builtin_amdgcn_wmma_f32_16x16x32_f16(
                    false, a, false, b, (short)0, acc[nt], false, false);
            }
        }

#pragma unroll
        for (int nt = 0; nt < 8; ++nt) {
            const int n = nt * 16 + m;
            const float bias = b1[n];
#pragma unroll
            for (int r = 0; r < 8; ++r) {
                float x = acc[nt][r] + bias;
                hTile[(r + 8 * h) * 128 + n] = (_Float16)silu_f(x);
            }
        }

        v8f acc2[8];
#pragma unroll
        for (int nt = 0; nt < 8; ++nt) acc2[nt] = vzero;

        for (int kc = 0; kc < 4; ++kc) {
            const _Float16* pa = hTile + m * 128 + kc * 32 + h * 8;
            v8h a0 = *(const v8h*)pa;
            v8h a1 = *(const v8h*)(pa + 16);
            v16h a;
#pragma unroll
            for (int i = 0; i < 8; ++i) { a[i] = a0[i]; a[i + 8] = a1[i]; }
            const _Float16* wbase = ldsW2 + kc * 8 * 512 + lane * 16;
#pragma unroll
            for (int nt = 0; nt < 8; ++nt) {
                v16h b = lds_b_frag(wbase + nt * 512);
                acc2[nt] = __builtin_amdgcn_wmma_f32_16x16x32_f16(
                    false, a, false, b, (short)0, acc2[nt], false, false);
            }
        }

        // bias (no activation on final layer), write f32 output
#pragma unroll
        for (int nt = 0; nt < 8; ++nt) {
            const int n = nt * 16 + m;
            const float bias = b2[n];
#pragma unroll
            for (int r = 0; r < 8; ++r) {
                const long long nr = (long long)tile * 16 + r + 8 * h;
                out[nr * NODE_C + n] = acc2[nt][r] + bias;
            }
        }
    }
}

// ---------------------------------------------------------------------------
extern "C" void kernel_launch(void* const* d_in, const int* in_sizes, int n_in,
                              void* d_out, int out_size, void* d_ws, size_t ws_size,
                              hipStream_t stream) {
    const float*     x_nodes = (const float*)d_in[0];
    const float*     x_edges = (const float*)d_in[1];
    const long long* ei      = (const long long*)d_in[2];
    const float* eW1 = (const float*)d_in[3];
    const float* eb1 = (const float*)d_in[4];
    const float* eW2 = (const float*)d_in[5];
    const float* eb2 = (const float*)d_in[6];
    const float* nW1 = (const float*)d_in[7];
    const float* nb1 = (const float*)d_in[8];
    const float* nW2 = (const float*)d_in[9];
    const float* nb2 = (const float*)d_in[10];

    const long long* sender   = ei;
    const long long* receiver = ei + N_EDGES;

    char* ws = (char*)d_ws;
    float* aggr = (float*)ws;
    size_t off = (size_t)N_NODES * NODE_C * sizeof(float);   // 25.6 MB
    _Float16* ewf = (_Float16*)(ws + off);                   // eW1|eW2 contiguous
    off += EDGE_W_BYTES;
    _Float16* nwf = (_Float16*)(ws + off);                   // nW1|nW2 contiguous
    off += NODE_W_BYTES;
    _Float16* ew2f = ewf + 320 * 128;
    _Float16* nw2f = nwf + 256 * 128;

    // allow >64KB dynamic LDS (WGP has 320KB)
    (void)hipFuncSetAttribute((const void*)edge_mlp_kernel,
                              hipFuncAttributeMaxDynamicSharedMemorySize,
                              EDGE_W_BYTES + HTILE_BYTES);
    (void)hipFuncSetAttribute((const void*)node_mlp_kernel,
                              hipFuncAttributeMaxDynamicSharedMemorySize,
                              NODE_W_BYTES + HTILE_BYTES);

    // 1) zero the aggregation buffer
    zero_f32_kernel<<<2048, 256, 0, stream>>>(aggr,
                                              (long long)N_NODES * NODE_C);

    // 2) swizzle weights into WMMA B-fragment layout (f32 -> f16)
    swizzle_weight_kernel<<<(320 * 128 + 255) / 256, 256, 0, stream>>>(eW1, ewf, 320, 128);
    swizzle_weight_kernel<<<(128 * 128 + 255) / 256, 256, 0, stream>>>(eW2, ew2f, 128, 128);
    swizzle_weight_kernel<<<(256 * 128 + 255) / 256, 256, 0, stream>>>(nW1, nwf, 256, 128);
    swizzle_weight_kernel<<<(128 * 128 + 255) / 256, 256, 0, stream>>>(nW2, nw2f, 128, 128);

    // 3) edge MLP + scatter-add: 50000 tiles, persistent blocks (~49 tiles ea)
    const int edgeTiles = N_EDGES / 16;
    edge_mlp_kernel<<<1024, 256, EDGE_W_BYTES + HTILE_BYTES, stream>>>(
        x_nodes, x_edges, sender, receiver, ewf, eb1, eb2, aggr, edgeTiles);

    // 4) node MLP: 3125 tiles, persistent blocks
    const int nodeTiles = (N_NODES + 15) / 16;
    node_mlp_kernel<<<256, 256, NODE_W_BYTES + HTILE_BYTES, stream>>>(
        x_nodes, aggr, nwf, nb1, nb2, (float*)d_out, nodeTiles);
}
